// GroupedKAAttention_58016418234718
// MI455X (gfx1250) — matrix operand order repaired and checked by reading, not compiled, not run on hardware
//
#include <hip/hip_runtime.h>
#include <hip/hip_bf16.h>

// ---------------------------------------------------------------------------
// Generic tiled GEMM:  C[m, n*ncs] = silu( sum_k A[m,k] * Bw[k,n] + bias[n] )
// fp32 in HBM -> native-cvt bf16 (v_cvt_pk_bf16_f32) -> double-buffered LDS ->
// v_wmma_f32_16x16x32_bf16 (fp32 accumulation).
// REQUIRES: M % 128 == 0, N % 128 == 0 (true for all launches below).
// K may be arbitrary (tail tile is the only guarded path).
// Block = 256 threads = 8 wave32; tile 128x128x32; wave tile 64x32.
// ---------------------------------------------------------------------------

#define BM 128
#define BN 128
#define BK 32
#define BKP 40   // padded LDS pitch (bf16 elements) to avoid bank conflicts

typedef float v8f __attribute__((ext_vector_type(8)));
typedef __attribute__((ext_vector_type(16))) __bf16 v16bf;
typedef __attribute__((ext_vector_type(8)))  __bf16 bf8;   // 16-byte chunk
typedef __attribute__((ext_vector_type(4)))  __bf16 bf4;   // 8-byte chunk

static __device__ __forceinline__ float silu_f(float x) {
  return x / (1.0f + __expf(-x));
}

// ---- register-staged tile loads (GUARD only instantiated for the K tail) ----
template <bool GUARD>
static __device__ __forceinline__ void load_a_regs(
    const float* __restrict__ Ab, int lda, int row, int kbase, int K, float v[16]) {
  const float* arow = Ab + (size_t)row * (size_t)lda + kbase;
  if (!GUARD) {
    const float4* p = (const float4*)arow;
    float4 q0 = p[0], q1 = p[1], q2 = p[2], q3 = p[3];
    v[0]=q0.x;  v[1]=q0.y;  v[2]=q0.z;  v[3]=q0.w;
    v[4]=q1.x;  v[5]=q1.y;  v[6]=q1.z;  v[7]=q1.w;
    v[8]=q2.x;  v[9]=q2.y;  v[10]=q2.z; v[11]=q2.w;
    v[12]=q3.x; v[13]=q3.y; v[14]=q3.z; v[15]=q3.w;
  } else {
#pragma unroll
    for (int i = 0; i < 16; i++) v[i] = (kbase + i < K) ? arow[i] : 0.0f;
  }
}

// B loader: thread owns 4 consecutive k-rows (bk4) x 4 columns (bc0).
template <bool GUARD>
static __device__ __forceinline__ void load_b_regs(
    const float* __restrict__ Bb, int ldb, int bn0, int bc0, int bk4,
    int k0, int K, float4 w[4]) {
#pragma unroll
  for (int r = 0; r < 4; r++) {
    const int kg = k0 + bk4 + r;
    if (!GUARD) {
      w[r] = *(const float4*)(Bb + (size_t)kg * (size_t)ldb + bn0 + bc0);
    } else {
      w[r] = make_float4(0.f, 0.f, 0.f, 0.f);
      if (kg < K) w[r] = *(const float4*)(Bb + (size_t)kg * (size_t)ldb + bn0 + bc0);
    }
  }
}

static __device__ __forceinline__ void store_a_lds(
    __bf16 (*As)[BKP], int ar, int ah, const float v[16]) {
  bf8 p0, p1;
#pragma unroll
  for (int i = 0; i < 8; i++) {
    p0[i] = (__bf16)v[i];        // native f32 -> bf16 cvt (packed)
    p1[i] = (__bf16)v[8 + i];
  }
  *(bf8*)&As[ar][ah * 16]     = p0;
  *(bf8*)&As[ar][ah * 16 + 8] = p1;
}

// Transposed store: for each of 4 columns, the 4 k-values are contiguous in
// Bs[n][k] -> packed cvt pairs + one ds_store_b64 per column.
static __device__ __forceinline__ void store_b_lds(
    __bf16 (*Bs)[BKP], int bc0, int bk4, const float4 w[4]) {
  bf4 v0 = {(__bf16)w[0].x, (__bf16)w[1].x, (__bf16)w[2].x, (__bf16)w[3].x};
  bf4 v1 = {(__bf16)w[0].y, (__bf16)w[1].y, (__bf16)w[2].y, (__bf16)w[3].y};
  bf4 v2 = {(__bf16)w[0].z, (__bf16)w[1].z, (__bf16)w[2].z, (__bf16)w[3].z};
  bf4 v3 = {(__bf16)w[0].w, (__bf16)w[1].w, (__bf16)w[2].w, (__bf16)w[3].w};
  *(bf4*)&Bs[bc0 + 0][bk4] = v0;
  *(bf4*)&Bs[bc0 + 1][bk4] = v1;
  *(bf4*)&Bs[bc0 + 2][bk4] = v2;
  *(bf4*)&Bs[bc0 + 3][bk4] = v3;
}

union FB { bf8 h[2]; v16bf v; };

// 8 WMMAs on one staged tile; all fragments loaded up front so the WMMA burst
// is not serialized on ds waits.
static __device__ __forceinline__ void compute_tile(
    const __bf16 (*Asb)[BKP], const __bf16 (*Bsb)[BKP],
    v8f acc[4][2], int lane, int wm, int wn) {
  const int ka = (lane < 16) ? 0 : 8;   // A/B frag: K 0-7/16-23 vs 8-15/24-31
  const int nlo = lane & 15;
  FB bf0, bf1, af[4];
  const int c0 = wn * 32 + nlo;
  bf0.h[0] = *(const bf8*)&Bsb[c0][ka];
  bf0.h[1] = *(const bf8*)&Bsb[c0][16 + ka];
  bf1.h[0] = *(const bf8*)&Bsb[c0 + 16][ka];
  bf1.h[1] = *(const bf8*)&Bsb[c0 + 16][16 + ka];
#pragma unroll
  for (int mt = 0; mt < 4; mt++) {
    const int row = wm * 64 + mt * 16 + nlo;
    af[mt].h[0] = *(const bf8*)&Asb[row][ka];
    af[mt].h[1] = *(const bf8*)&Asb[row][16 + ka];
  }
#pragma unroll
  for (int mt = 0; mt < 4; mt++) {
    acc[mt][0] = __builtin_amdgcn_wmma_f32_16x16x32_bf16(
        false, af[mt].v, false, bf0.v, (short)0, acc[mt][0], false, false);
    acc[mt][1] = __builtin_amdgcn_wmma_f32_16x16x32_bf16(
        false, af[mt].v, false, bf1.v, (short)0, acc[mt][1], false, false);
  }
}

__global__ __launch_bounds__(256) void gemm_bf16_silu_wmma(
    const float* __restrict__ A,  long long zA,  int lda,
    const float* __restrict__ Bw, long long zB,  int ldb,
    const float* __restrict__ bias, long long zBias,
    float* __restrict__ C, long long zC, int ldc, int ncs,
    int N, int K)
{
  const int g = blockIdx.z;
  const float* Ab    = A    + (size_t)g * (size_t)zA;
  const float* Bb    = Bw   + (size_t)g * (size_t)zB;
  const float* biasb = bias + (size_t)g * (size_t)zBias;
  float*       Cb    = C    + (size_t)g * (size_t)zC;

  const int bn0 = blockIdx.x * BN;
  const int bm0 = blockIdx.y * BM;

  __shared__ alignas(16) __bf16 As[2][BM][BKP];  // [m][k] bf16
  __shared__ alignas(16) __bf16 Bs[2][BN][BKP];  // [n][k] bf16 (B^T)

  const int tid  = threadIdx.x;
  const int lane = tid & 31;
  const int wave = tid >> 5;
  const int wm   = wave & 1;   // 0..1  -> 64 rows each
  const int wn   = wave >> 1;  // 0..3  -> 32 cols each

  v8f acc[4][2];
#pragma unroll
  for (int i = 0; i < 4; i++)
#pragma unroll
    for (int j = 0; j < 2; j++)
#pragma unroll
      for (int v = 0; v < 8; v++) acc[i][j][v] = 0.0f;

  // A loader mapping: thread -> (row, 16-wide half of the 32-wide k slab)
  const int ar = tid >> 1;         // 0..127
  const int ah = tid & 1;          // 0..1
  // B loader mapping: thread -> (4 consecutive k rows, 4-col chunk)
  const int bk4 = (tid >> 5) * 4;  // 0,4,...,28
  const int bc0 = (tid & 31) * 4;  // 0..124

  const int ntiles = (K + BK - 1) / BK;
  const int arow_g = bm0 + ar;

  float  av[16];
  float4 bw4[4];

  // ---- prologue: stage tile 0 ----
  if (BK > K) {
    load_a_regs<true >(Ab, lda, arow_g, ah * 16, K, av);
    load_b_regs<true >(Bb, ldb, bn0, bc0, bk4, 0, K, bw4);
  } else {
    load_a_regs<false>(Ab, lda, arow_g, ah * 16, K, av);
    load_b_regs<false>(Bb, ldb, bn0, bc0, bk4, 0, K, bw4);
  }
  store_a_lds(As[0], ar, ah, av);
  store_b_lds(Bs[0], bc0, bk4, bw4);

  // ---- pipelined main loop: load t+1 | wmma t | store t+1 ----
  for (int t = 0; t < ntiles; ++t) {
    __syncthreads();
    const bool more = (t + 1 < ntiles);
    if (more) {
      const int k0n = (t + 1) * BK;
      if (k0n + BK > K) {   // tail tile (only possible for the last tile)
        load_a_regs<true >(Ab, lda, arow_g, k0n + ah * 16, K, av);
        load_b_regs<true >(Bb, ldb, bn0, bc0, bk4, k0n, K, bw4);
      } else {
        load_a_regs<false>(Ab, lda, arow_g, k0n + ah * 16, K, av);
        load_b_regs<false>(Bb, ldb, bn0, bc0, bk4, k0n, K, bw4);
      }
      if (t + 2 < ntiles)   // GL2 prefetch of the tile after next (weights)
        __builtin_prefetch(Bb + (size_t)((t + 2) * BK + bk4) * (size_t)ldb + bn0 + bc0, 0, 1);
    }
    compute_tile(As[t & 1], Bs[t & 1], acc, lane, wm, wn);
    if (more) {
      store_a_lds(As[(t + 1) & 1], ar, ah, av);
      store_b_lds(Bs[(t + 1) & 1], bc0, bk4, bw4);
    }
  }

  // ---- epilogue: bias + SiLU, strided store (ncs = output column stride) ----
  const int nlo = lane & 15;
  const int mhi = (lane >> 4) * 8;   // C layout: lanes 16-31 hold M = v+8
#pragma unroll
  for (int mt = 0; mt < 4; mt++) {
#pragma unroll
    for (int nt = 0; nt < 2; nt++) {
      const int n = bn0 + wn * 32 + nt * 16 + nlo;
      const float bval = biasb[n];
#pragma unroll
      for (int v = 0; v < 8; v++) {
        const int m = bm0 + wm * 64 + mt * 16 + v + mhi;
        const float x = acc[mt][nt][v] + bval;
        Cb[(size_t)m * (size_t)ldc + (size_t)n * (size_t)ncs] = silu_f(x);
      }
    }
  }
}

// ---------------------------------------------------------------------------
// scores[b] = dot(qout[b], kout[b]); out = softmax over the 256-row batch
// ---------------------------------------------------------------------------
__global__ __launch_bounds__(256) void dot_softmax_kernel(
    const float* __restrict__ qo, const float* __restrict__ ko,
    float* __restrict__ out, int D)
{
  __shared__ float red[256];
  const int b = threadIdx.x;
  const float4* qr = (const float4*)(qo + (size_t)b * D);
  const float4* kr = (const float4*)(ko + (size_t)b * D);
  float s = 0.0f;
  for (int i = 0; i < D / 4; i++) {
    float4 a = qr[i], c = kr[i];
    s += a.x * c.x + a.y * c.y + a.z * c.z + a.w * c.w;
  }
  red[b] = s;
  __syncthreads();
  for (int off = 128; off > 0; off >>= 1) {
    if (b < off) red[b] = fmaxf(red[b], red[b + off]);
    __syncthreads();
  }
  const float mx = red[0];
  __syncthreads();
  const float e = __expf(s - mx);
  red[b] = e;
  __syncthreads();
  for (int off = 128; off > 0; off >>= 1) {
    if (b < off) red[b] += red[b + off];
    __syncthreads();
  }
  out[b] = e / red[0];
}

// ---------------------------------------------------------------------------
extern "C" void kernel_launch(void* const* d_in, const int* in_sizes, int n_in,
                              void* d_out, int out_size, void* d_ws, size_t ws_size,
                              hipStream_t stream) {
  const float* q   = (const float*)d_in[0];
  const float* k   = (const float*)d_in[1];
  const float* Wq1 = (const float*)d_in[2];
  const float* bq1 = (const float*)d_in[3];
  const float* Wq4 = (const float*)d_in[4];
  const float* bq4 = (const float*)d_in[5];
  const float* Wk1 = (const float*)d_in[6];
  const float* bk1 = (const float*)d_in[7];
  const float* Wk4 = (const float*)d_in[8];
  const float* bk4 = (const float*)d_in[9];
  const float* Wg1 = (const float*)d_in[10];
  const float* bg1 = (const float*)d_in[11];
  const float* Wg4 = (const float*)d_in[12];
  const float* bg4 = (const float*)d_in[13];

  constexpr int Bsz = 256, G = 16, GS = 588, HID = 1024, FEAT = 2048;

  // Workspace layout (floats). k-branch reuses H/F; peak ~57 MB.
  float* ws  = (float*)d_ws;
  float* H   = ws;                                   // [B, G*HID]
  float* F   = H   + (size_t)Bsz * G * HID;          // [B, G*FEAT] (interleaved)
  float* qHg = F   + (size_t)Bsz * G * FEAT;         // [B, HID]
  float* kHg = qHg + (size_t)Bsz * HID;
  float* qO  = kHg + (size_t)Bsz * HID;              // [B, FEAT]
  float* kO  = qO  + (size_t)Bsz * FEAT;

  const float* X [2] = {q,   k  };
  const float* W1[2] = {Wq1, Wk1};
  const float* B1[2] = {bq1, bk1};
  const float* W4[2] = {Wq4, Wk4};
  const float* B4[2] = {bq4, bk4};
  float*       Hg[2] = {qHg, kHg};
  float*       O [2] = {qO,  kO };

  dim3 blk(256);
  for (int br = 0; br < 2; br++) {
    // stage 1: per-group fc1  [256x588]x[588x1024] x16 groups -> H[b, g*HID+h]
    gemm_bf16_silu_wmma<<<dim3(HID / 128, Bsz / 128, G), blk, 0, stream>>>(
        X[br], (long long)GS, G * GS,
        W1[br], (long long)GS * HID, HID,
        B1[br], (long long)HID,
        H, (long long)HID, G * HID, 1,
        HID, GS);
    // stage 2: per-group fc4  [256x1024]x[1024x2048] x16 -> F[b, o*G + g]
    gemm_bf16_silu_wmma<<<dim3(FEAT / 128, Bsz / 128, G), blk, 0, stream>>>(
        H, (long long)HID, G * HID,
        W4[br], (long long)HID * FEAT, FEAT,
        B4[br], (long long)FEAT,
        F, 1LL, G * FEAT, G,
        FEAT, HID);
    // stage 3: global fc1  [256x32768]x[32768x1024] -> Hg
    gemm_bf16_silu_wmma<<<dim3(HID / 128, Bsz / 128, 1), blk, 0, stream>>>(
        F, 0LL, G * FEAT,
        Wg1, 0LL, HID,
        bg1, 0LL,
        Hg[br], 0LL, HID, 1,
        HID, G * FEAT);
    // stage 4: global fc4  [256x1024]x[1024x2048] -> O
    gemm_bf16_silu_wmma<<<dim3(FEAT / 128, Bsz / 128, 1), blk, 0, stream>>>(
        Hg[br], 0LL, HID,
        Wg4, 0LL, FEAT,
        bg4, 0LL,
        O[br], 0LL, FEAT, 1,
        FEAT, HID);
  }
  // stage 5: per-row dot + softmax over batch
  dot_softmax_kernel<<<1, blk, 0, stream>>>(qO, kO, (float*)d_out, FEAT);
}
